// LocalDecoder_64888365908194
// MI455X (gfx1250) — matrix operand order repaired and checked by reading, not compiled, not run on hardware
//
#include <hip/hip_runtime.h>
#include <hip/hip_bf16.h>

typedef __attribute__((ext_vector_type(16))) _Float16 v16h;
typedef __attribute__((ext_vector_type(8)))  float    v8f;
typedef __attribute__((ext_vector_type(4)))  unsigned int v4u;

union AFrag   { v16h h; v4u u[2]; };
union PackF16 { _Float16 h[2]; unsigned int u; };

#define NEDGES   200000
// workspace layout (dword offsets)
#define OFF_W1   0        // 2*5*4*256  = 10240 dwords (f16-pair B fragments, 160x64 per feat)
#define OFF_W2   10240    // 2*2*4*256  =  4096 dwords
#define OFF_W3   14336    // 2*2*1*256  =  1024 dwords (only column 0 nonzero)
#define OFF_C1   15360    // 2*64 floats (b1 + folded global contributions)
#define OFF_B2   15488    // 2*64 floats
#define OFF_B3   15616    // 2 floats
#define WS_DWORDS 15618

// ---------------------------------------------------------------------------
// Prep: pack f32 weights into f16 WMMA B-fragment layout; fold globals into c1
// B packing chosen here (contiguous k, same as A reads): lane l -> n=l&15, half=l>>4; dword r holds
// K = half*16 + 2r (lo) and +1 (hi).
// ---------------------------------------------------------------------------
__global__ void prep_kernel(const float* __restrict__ W1, const float* __restrict__ b1,
                            const float* __restrict__ W2, const float* __restrict__ b2_,
                            const float* __restrict__ W3, const float* __restrict__ b3_,
                            const float* __restrict__ h_global,
                            const float* __restrict__ enc_global,
                            unsigned int* __restrict__ ws) {
  int idx = blockIdx.x * blockDim.x + threadIdx.x;
  float* wsf = (float*)ws;
  if (idx < 10240) {                       // W1 fragments, K reduced 256->160
    int r = idx % 8;  int t1 = idx / 8;
    int lane = t1 % 32; int t2 = t1 / 32;
    int j = t2 % 4;   int t3 = t2 / 4;
    int kt = t3 % 5;  int f = t3 / 5;
    int n = j * 16 + (lane & 15);
    int halfv = lane >> 4;
    int kl = kt * 32 + halfv * 16 + 2 * r;          // 0..159
    int k0 = (kl < 128) ? kl : (kl + 64);           // skip h_global rows [128,192)
    PackF16 p;
    p.h[0] = (_Float16)W1[(f * 256 + k0) * 64 + n];
    p.h[1] = (_Float16)W1[(f * 256 + k0 + 1) * 64 + n];
    ws[OFF_W1 + idx] = p.u;
  } else if (idx < 14336) {                // W2 fragments
    int i = idx - 10240;
    int r = i % 8;  int t1 = i / 8;
    int lane = t1 % 32; int t2 = t1 / 32;
    int j = t2 % 4;  int t3 = t2 / 4;
    int s = t3 % 2;  int f = t3 / 2;
    int n = j * 16 + (lane & 15);
    int halfv = lane >> 4;
    int kl = s * 32 + halfv * 16 + 2 * r;
    PackF16 p;
    p.h[0] = (_Float16)W2[(f * 64 + kl) * 64 + n];
    p.h[1] = (_Float16)W2[(f * 64 + kl + 1) * 64 + n];
    ws[OFF_W2 + i] = p.u;
  } else if (idx < 15360) {                // W3 fragments (column 0 only)
    int i = idx - 14336;
    int r = i % 8;  int t1 = i / 8;
    int lane = t1 % 32; int t2 = t1 / 32;
    int s = t2 % 2;  int f = t2 / 2;
    int nl = lane & 15;
    int halfv = lane >> 4;
    int kl = s * 32 + halfv * 16 + 2 * r;
    PackF16 p;
    p.h[0] = (nl == 0) ? (_Float16)W3[f * 64 + kl]     : (_Float16)0.0f;
    p.h[1] = (nl == 0) ? (_Float16)W3[f * 64 + kl + 1] : (_Float16)0.0f;
    ws[OFF_W3 + i] = p.u;
  } else if (idx < 15488) {                // c1 = b1 + globals . W1[global rows]
    int i = idx - 15360;
    int f = i / 64, n = i % 64;
    float acc = b1[f * 64 + n];
    for (int k = 0; k < 64; ++k) acc += h_global[k]   * W1[(f * 256 + 128 + k) * 64 + n];
    for (int k = 0; k < 32; ++k) acc += enc_global[k] * W1[(f * 256 + 224 + k) * 64 + n];
    wsf[OFF_C1 + i] = acc;
  } else if (idx < 15616) {
    int i = idx - 15488;
    wsf[OFF_B2 + i] = b2_[i];
  } else if (idx < WS_DWORDS) {
    int i = idx - 15616;
    wsf[OFF_B3 + i] = b3_[i];
  }
}

// ---------------------------------------------------------------------------
// Main: 8 waves/WG, 4 edge-tiles of 16 x 2 features. WMMA f16->f32 MLP.
// ---------------------------------------------------------------------------
__global__ __launch_bounds__(256) void decoder_kernel(
    const float* __restrict__ edge_array, const float* __restrict__ h_local,
    const float* __restrict__ enc_local,  const int* __restrict__ addr_from,
    const int* __restrict__ addr_to,      const unsigned int* __restrict__ ws,
    float* __restrict__ out) {
  // X: 4 tiles x 16 edges x 160 f16, row stride 84 dwords (bank-conflict-free)
  __shared__ __align__(16) unsigned int Xlds[4][16 * 84];
  // per-wave H staging: 16 x 64 f16, row stride 72 f16 (36 dwords)
  __shared__ __align__(16) _Float16 Hs[8][2][16 * 72];

  const int tid = threadIdx.x;
  const int wgBase = blockIdx.x * 64;

  // ---- cooperative gather + f32->f16 pack of X tiles (64 edges x 80 dwords)
  for (int it = 0; it < 20; ++it) {
    int idx  = it * 256 + tid;        // < 5120
    int eloc = idx / 80;              // 0..63
    int c    = idx % 80;              // dword column (2 f16)
    int e    = wgBase + eloc;
    float f0, f1;
    if (c < 32) {
      int a = addr_from[e];
      const float* p = h_local + a * 64 + c * 2;
      f0 = p[0]; f1 = p[1];
    } else if (c < 64) {
      int a = addr_to[e];
      const float* p = h_local + a * 64 + (c - 32) * 2;
      f0 = p[0]; f1 = p[1];
    } else {
      const float* p = enc_local + e * 32 + (c - 64) * 2;
      f0 = p[0]; f1 = p[1];
    }
    PackF16 pk; pk.h[0] = (_Float16)f0; pk.h[1] = (_Float16)f1;
    Xlds[eloc >> 4][(eloc & 15) * 84 + c] = pk.u;
  }
  __syncthreads();

  const int wave = tid >> 5, lane = tid & 31;
  const int t = wave >> 1, f = wave & 1;       // tile, feature
  const int nl = lane & 15, halfv = lane >> 4;
  const int m = nl;                            // A-fragment row
  const unsigned int* Xt = &Xlds[t][0];
  const float* wsf = (const float*)ws;

  // ---- layer 1: [16x160] x [160x64], C init = c1 (b1 + folded globals)
  v8f acc[4];
#pragma unroll
  for (int j = 0; j < 4; ++j) {
    float cv = wsf[OFF_C1 + f * 64 + j * 16 + nl];
#pragma unroll
    for (int r = 0; r < 8; ++r) acc[j][r] = cv;
  }
#pragma unroll
  for (int s = 0; s < 5; ++s) {
    AFrag a;
    a.u[0] = *(const v4u*)&Xt[m * 84 + s * 16 + halfv * 8];
    a.u[1] = *(const v4u*)&Xt[m * 84 + s * 16 + halfv * 8 + 4];
#pragma unroll
    for (int j = 0; j < 4; ++j) {
      v16h b = ((const v16h*)(ws + OFF_W1 + ((f * 5 + s) * 4 + j) * 256))[lane];
      acc[j] = __builtin_amdgcn_wmma_f32_16x16x32_f16(false, a.h, false, b,
                                                      (short)0, acc[j], false, false);
    }
  }
  // leaky-relu + f16 store to wave-private staging (C layout: M = halfv*8 + r)
  _Float16* H1 = &Hs[wave][0][0];
#pragma unroll
  for (int j = 0; j < 4; ++j)
#pragma unroll
    for (int r = 0; r < 8; ++r) {
      float v = acc[j][r];
      v = fmaxf(v, 0.01f * v);
      H1[(halfv * 8 + r) * 72 + j * 16 + nl] = (_Float16)v;
    }
  asm volatile("s_wait_dscnt 0" ::: "memory");   // wave-local LDS RAW

  // ---- layer 2: [16x64] x [64x64]
  v8f acc2[4];
#pragma unroll
  for (int j = 0; j < 4; ++j) {
    float bv = wsf[OFF_B2 + f * 64 + j * 16 + nl];
#pragma unroll
    for (int r = 0; r < 8; ++r) acc2[j][r] = bv;
  }
  const unsigned int* H1d = (const unsigned int*)H1;
#pragma unroll
  for (int s = 0; s < 2; ++s) {
    AFrag a;
    a.u[0] = *(const v4u*)&H1d[m * 36 + s * 16 + halfv * 8];
    a.u[1] = *(const v4u*)&H1d[m * 36 + s * 16 + halfv * 8 + 4];
#pragma unroll
    for (int j = 0; j < 4; ++j) {
      v16h b = ((const v16h*)(ws + OFF_W2 + ((f * 2 + s) * 4 + j) * 256))[lane];
      acc2[j] = __builtin_amdgcn_wmma_f32_16x16x32_f16(false, a.h, false, b,
                                                       (short)0, acc2[j], false, false);
    }
  }
  _Float16* H2 = &Hs[wave][1][0];
#pragma unroll
  for (int j = 0; j < 4; ++j)
#pragma unroll
    for (int r = 0; r < 8; ++r) {
      float v = acc2[j][r];
      v = fmaxf(v, 0.01f * v);
      H2[(halfv * 8 + r) * 72 + j * 16 + nl] = (_Float16)v;
    }
  asm volatile("s_wait_dscnt 0" ::: "memory");

  // ---- layer 3: [16x64] x [64x1] via W3 fragment with only column 0 nonzero
  v8f acc3;
  {
    float bv = (nl == 0) ? wsf[OFF_B3 + f] : 0.0f;
#pragma unroll
    for (int r = 0; r < 8; ++r) acc3[r] = bv;
  }
  const unsigned int* H2d = (const unsigned int*)H2;
#pragma unroll
  for (int s = 0; s < 2; ++s) {
    AFrag a;
    a.u[0] = *(const v4u*)&H2d[m * 36 + s * 16 + halfv * 8];
    a.u[1] = *(const v4u*)&H2d[m * 36 + s * 16 + halfv * 8 + 4];
    v16h b = ((const v16h*)(ws + OFF_W3 + (f * 2 + s) * 256))[lane];
    acc3 = __builtin_amdgcn_wmma_f32_16x16x32_f16(false, a.h, false, b,
                                                  (short)0, acc3, false, false);
  }
  // column 0 lives in lanes 0 (M=r) and 16 (M=8+r)
  if (nl == 0) {
    int eBase = (blockIdx.x * 4 + t) * 16;
#pragma unroll
    for (int r = 0; r < 8; ++r) {
      int mm = halfv * 8 + r;
      int e  = eBase + mm;
      float v = acc3[r];
      float e0 = edge_array[4 * e];
      if (e0 != e0) v = __builtin_nanf("");
      out[f * NEDGES + e] = v;
    }
  }
}

extern "C" void kernel_launch(void* const* d_in, const int* in_sizes, int n_in,
                              void* d_out, int out_size, void* d_ws, size_t ws_size,
                              hipStream_t stream) {
  const float* edge_array = (const float*)d_in[0];
  const float* h_local    = (const float*)d_in[1];
  const float* h_global   = (const float*)d_in[2];
  const float* enc_local  = (const float*)d_in[3];
  const float* enc_global = (const float*)d_in[4];
  const float* W1 = (const float*)d_in[5];
  const float* b1 = (const float*)d_in[6];
  const float* W2 = (const float*)d_in[7];
  const float* b2 = (const float*)d_in[8];
  const float* W3 = (const float*)d_in[9];
  const float* b3 = (const float*)d_in[10];
  const int* addr_from = (const int*)d_in[11];
  const int* addr_to   = (const int*)d_in[12];
  float* out = (float*)d_out;
  unsigned int* ws = (unsigned int*)d_ws;

  prep_kernel<<<(WS_DWORDS + 255) / 256, 256, 0, stream>>>(
      W1, b1, W2, b2, W3, b3, h_global, enc_global, ws);

  decoder_kernel<<<NEDGES / 64, 256, 0, stream>>>(
      edge_array, h_local, enc_local, addr_from, addr_to, ws, out);
}
